// SSA_64587718197956
// MI455X (gfx1250) — matrix operand order repaired and checked by reading, not compiled, not run on hardware
//
#include <hip/hip_runtime.h>
#include <hip/hip_bf16.h>

// ---------------------------------------------------------------------------
// Spiking Self-Attention block for MI455X (gfx1250), wave32 + WMMA f16 +
// async global->LDS weight staging (ASYNCcnt / s_wait_asynccnt).
// Spikes are exactly representable in f16; attention math is exact.
// ---------------------------------------------------------------------------

typedef __attribute__((ext_vector_type(16))) _Float16 v16h;
typedef __attribute__((ext_vector_type(8)))  _Float16 v8h;
typedef __attribute__((ext_vector_type(8)))  float    v8f;
typedef __attribute__((ext_vector_type(4)))  float    v4f;

#define TT 4
#define BB 16
#define CC 768
#define HH 16
#define WW 16
#define NN 256            // H*W
#define TBX 64            // T*B
#define NHD 8
#define DH 96
#define KSTEPS 216        // 9 taps * (768/32) ci-blocks
#define BCN ((size_t)BB * CC * NN)   // elements per timestep

__device__ __forceinline__ v8f wmma16(v16h a, v16h b, v8f c) {
  return __builtin_amdgcn_wmma_f32_16x16x32_f16(
      /*neg_a=*/false, a, /*neg_b=*/false, b,
      /*c_mod=*/(short)0, c, /*reuse_a=*/false, /*reuse_b=*/false);
}

// Per-lane A fragment (16x32 f16): K = kb..kb+15, kb = 16*(lane>=16).
__device__ __forceinline__ v16h load_a_frag(const _Float16* p, int kb) {
  return *(const v16h*)(p + kb);
}

// Async DMA: 16B global -> LDS, tracked with ASYNCcnt (ISA 08_async_tensor §4).
__device__ __forceinline__ void async_copy_b128(unsigned lds_off, const void* gptr) {
  asm volatile("global_load_async_to_lds_b128 %0, %1, off"
               :: "v"(lds_off), "v"(gptr) : "memory");
}
__device__ __forceinline__ void wait_async0() {
  asm volatile("s_wait_asynccnt 0x0" ::: "memory");
}

// ---------------------------------------------------------------------------
// BN folding: scale = gamma/sqrt(var+eps), bias = beta - mean*scale
// ---------------------------------------------------------------------------
__global__ void bnprep(const float* __restrict__ g, const float* __restrict__ be,
                       const float* __restrict__ mn, const float* __restrict__ vr,
                       float* __restrict__ scale, float* __restrict__ bias) {
  int c = blockIdx.x * 256 + threadIdx.x;
  if (c < CC) {
    float s = g[c] * rsqrtf(vr[c] + 1e-5f);
    scale[c] = s;
    bias[c]  = be[c] - mn[c] * s;
  }
}

// ---------------------------------------------------------------------------
// Weight transform: w (C,C,3,3) f32 -> wt (9,C,C) f16, wt[tap][co][ci]
// ---------------------------------------------------------------------------
__global__ void wtrans(const float* __restrict__ w, _Float16* __restrict__ wt) {
  int idx = blockIdx.x * 256 + threadIdx.x;
  if (idx >= CC * CC) return;
  int co = idx / CC, ci = idx % CC;
#pragma unroll
  for (int tap = 0; tap < 9; ++tap)
    wt[(size_t)tap * CC * CC + (size_t)co * CC + ci] =
        (_Float16)w[(size_t)idx * 9 + tap];
}

// ---------------------------------------------------------------------------
// LIF with layout transpose: src (T,B,C,N) f32 -> dst (TB,N,C) f16 spikes.
// ---------------------------------------------------------------------------
__global__ __launch_bounds__(256) void lif_transpose(
    const float* __restrict__ src, _Float16* __restrict__ dst, float vth) {
  __shared__ float tile[32][33];
  const int tid = threadIdx.x;
  const int cl  = tid & 31;
  const int rg  = tid >> 5;
  const int c0  = blockIdx.x * 32;
  const int n0  = blockIdx.y * 32;
  const int b   = blockIdx.z;
  float v[4] = {0.f, 0.f, 0.f, 0.f};
#pragma unroll
  for (int t = 0; t < TT; ++t) {
#pragma unroll
    for (int j = 0; j < 4; ++j)
      tile[rg + 8 * j][cl] =
          src[((size_t)(t * BB + b) * CC + c0 + rg + 8 * j) * NN + n0 + cl];
    __syncthreads();
#pragma unroll
    for (int j = 0; j < 4; ++j) {
      float x = tile[cl][rg + 8 * j];
      float h = v[j] + (x - v[j]) * 0.5f;
      bool  s = (h >= vth);
      v[j]    = s ? 0.f : h;
      dst[((size_t)(t * BB + b) * NN + n0 + rg + 8 * j) * CC + c0 + cl] =
          (_Float16)(s ? 1.f : 0.f);
    }
    __syncthreads();
  }
}

// ---------------------------------------------------------------------------
// LIF without transpose: src (T,B,C,N) f32 -> dst (TB,C,N) f16 spikes.
// ---------------------------------------------------------------------------
__global__ void lif_plain(const float* __restrict__ src,
                          _Float16* __restrict__ dst, float vth) {
  size_t idx = (size_t)blockIdx.x * 256 + threadIdx.x;
  float v = 0.f;
#pragma unroll
  for (int t = 0; t < TT; ++t) {
    float x = src[(size_t)t * BCN + idx];
    float h = v + (x - v) * 0.5f;
    bool  s = (h >= vth);
    v       = s ? 0.f : h;
    dst[(size_t)t * BCN + idx] = (_Float16)(s ? 1.f : 0.f);
  }
}

// ---------------------------------------------------------------------------
// 3x3 SAME conv + BN as implicit-im2col WMMA GEMM.
//   in  : (TB, H, W, C) f16 spikes          wt : (9, C, C) f16 [tap][co][ci]
//   out : (TB, C, N) f32, BN applied
// Block = 8 waves = 4 pixel-rows x 2 co-halves (64 px x 128 co).
// Weight tile (128 co x 32 k = 8KB) is async-DMA'd into double-buffered LDS
// one k-step ahead; WMMA consumes B fragments from LDS via ds_load_b128.
// ---------------------------------------------------------------------------
__global__ __launch_bounds__(256) void conv_bn_wmma(
    const _Float16* __restrict__ in, const _Float16* __restrict__ wt,
    const float* __restrict__ scale, const float* __restrict__ bias,
    float* __restrict__ out) {
  __shared__ __align__(16) _Float16 Bstage[2][128 * 32];  // [buf][co_local*32+k]
  const int tid  = threadIdx.x;
  const int lane = tid & 31;
  const int wave = tid >> 5;
  const int m    = lane & 15;
  const int hi   = lane >> 4;
  const int kbA  = hi * 16;
  const int kbB  = hi * 16;
  const int rw   = wave >> 1;                 // 0..3: row within block
  const int cw   = wave & 1;                  // 0..1: co half
  const int r    = blockIdx.y * 4 + rw;
  const int cob0 = blockIdx.x * 128;
  const int co_base = cob0 + cw * 64;
  const int tb   = blockIdx.z;

  const unsigned lds0 = (unsigned)(uintptr_t)&Bstage[0][0];

  // stage k-step `ks` weight tile into buffer `buf` (2x16B chunks per thread)
  auto stage = [&](int ks, int buf) {
    const int tap = ks / 24, cb = ks % 24;
    const _Float16* wbase =
        wt + (size_t)tap * CC * CC + (size_t)cob0 * CC + cb * 32;
    const unsigned lbase = lds0 + (unsigned)buf * (128 * 32 * 2);
#pragma unroll
    for (int s = 0; s < 2; ++s) {
      const int c   = tid + 256 * s;          // chunk 0..511
      const int row = c >> 2, q = c & 3;      // co row, 16B quarter
      async_copy_b128(lbase + (unsigned)c * 16,
                      (const void*)(wbase + (size_t)row * CC + q * 8));
    }
  };

  v8f acc[4] = {{}, {}, {}, {}};

  stage(0, 0);
  wait_async0();
  __syncthreads();

  for (int ks = 0; ks < KSTEPS; ++ks) {
    const int p = ks & 1;
    if (ks + 1 < KSTEPS) stage(ks + 1, p ^ 1);

    const int tap = ks / 24, cb = ks % 24;
    const int dy = tap / 3 - 1, dx = tap % 3 - 1;
    const int yy = r + dy;
    if (yy >= 0 && yy < HH) {                 // uniform per wave
      const int  xcol  = m + dx;
      const bool valid = (xcol >= 0) && (xcol < WW);
      const _Float16* arow =
          in + (((size_t)tb * HH + yy) * WW + (valid ? xcol : 0)) * CC + cb * 32;
      if (cb + 1 < 24) __builtin_prefetch(arow + 32, 0, 3);
      v16h a = load_a_frag(arow, kbA);        // branchless edge mask
      v16h z = {};
      a = valid ? a : z;
      const _Float16* bbuf = &Bstage[p][(cw * 64) * 32];
      v16h bf[4];
#pragma unroll
      for (int tq = 0; tq < 4; ++tq)
        bf[tq] = *(const v16h*)(bbuf + (tq * 16 + m) * 32 + kbB);
#pragma unroll
      for (int tq = 0; tq < 4; ++tq) acc[tq] = wmma16(a, bf[tq], acc[tq]);
    }
    wait_async0();      // next tile landed (this wave's DMA)
    __syncthreads();    // ...and everyone's; previous buffer free to overwrite
  }

#pragma unroll
  for (int tq = 0; tq < 4; ++tq) {
    const int co = co_base + tq * 16 + m;
    const float s = scale[co], bb = bias[co];
    float* op = out + ((size_t)tb * CC + co) * NN + r * 16 + hi * 8;
    v4f r0, r1;
#pragma unroll
    for (int j = 0; j < 4; ++j) {
      r0[j] = acc[tq][j] * s + bb;
      r1[j] = acc[tq][4 + j] * s + bb;
    }
    *(v4f*)op       = r0;
    *(v4f*)(op + 4) = r1;
  }
}

// ---------------------------------------------------------------------------
// Linear attention per (t,b,head): kv = k^T v (96x96, K=256), then
// o = q * kv * 0.125 (256x96, K=96). kv staged in LDS as f16 [e][d].
//   q_sp : (TB,N,C) f16    k_sp,v_sp : (TB,C,N) f16    o : (TB,C,N) f32
// ---------------------------------------------------------------------------
__global__ __launch_bounds__(256) void attn_wmma(
    const _Float16* __restrict__ q_sp, const _Float16* __restrict__ k_sp,
    const _Float16* __restrict__ v_sp, float* __restrict__ o) {
  __shared__ __align__(32) _Float16 kv[DH * DH];  // [e][d]
  const int lane = threadIdx.x & 31;
  const int wave = threadIdx.x >> 5;
  const int m    = lane & 15;
  const int hi   = lane >> 4;
  const int kbA  = hi * 16;
  const int kbB  = hi * 16;
  const int tb   = blockIdx.x >> 3;
  const int h    = blockIdx.x & 7;
  const size_t cbase = (size_t)tb * CC + h * DH;

  if (wave < 6) {                 // phase 1: kv[d][e] = sum_n k[n,d] v[n,e]
    const int d0 = wave * 16;
    v8f acc[6] = {};
    for (int nb = 0; nb < NN / 32; ++nb) {
      v16h a = load_a_frag(k_sp + (cbase + d0 + m) * NN + nb * 32, kbA);
#pragma unroll
      for (int et = 0; et < 6; ++et) {
        v16h b = *(const v16h*)(v_sp + (cbase + et * 16 + m) * NN + nb * 32 + kbB);
        acc[et] = wmma16(a, b, acc[et]);
      }
    }
#pragma unroll
    for (int et = 0; et < 6; ++et) {
      const int e = et * 16 + m;
      v8h st;
#pragma unroll
      for (int j = 0; j < 8; ++j) st[j] = (_Float16)acc[et][j];
      *(v8h*)&kv[e * DH + d0 + hi * 8] = st;
    }
  }
  __syncthreads();

  v8f acc2[2][6] = {};            // phase 2: o = 0.125 * q @ kv
#pragma unroll
  for (int db = 0; db < 3; ++db) {
    v16h aA[2];
#pragma unroll
    for (int pt = 0; pt < 2; ++pt) {
      const int pix = (wave * 2 + pt) * 16 + m;
      aA[pt] = load_a_frag(q_sp + ((size_t)tb * NN + pix) * CC + h * DH + db * 32, kbA);
    }
#pragma unroll
    for (int et = 0; et < 6; ++et) {
      v16h b = *(const v16h*)&kv[(et * 16 + m) * DH + db * 32 + kbB];
#pragma unroll
      for (int pt = 0; pt < 2; ++pt) acc2[pt][et] = wmma16(aA[pt], b, acc2[pt][et]);
    }
  }
#pragma unroll
  for (int pt = 0; pt < 2; ++pt) {
#pragma unroll
    for (int et = 0; et < 6; ++et) {
      const int e = et * 16 + m;
      float* op = o + (cbase + e) * NN + (wave * 2 + pt) * 16 + hi * 8;
      v4f r0, r1;
#pragma unroll
      for (int j = 0; j < 4; ++j) {
        r0[j] = acc2[pt][et][j] * 0.125f;
        r1[j] = acc2[pt][et][4 + j] * 0.125f;
      }
      *(v4f*)op       = r0;
      *(v4f*)(op + 4) = r1;
    }
  }
}

// ---------------------------------------------------------------------------
extern "C" void kernel_launch(void* const* d_in, const int* in_sizes, int n_in,
                              void* d_out, int out_size, void* d_ws, size_t ws_size,
                              hipStream_t stream) {
  (void)in_sizes; (void)n_in; (void)out_size; (void)ws_size;
  const float* x = (const float*)d_in[0];

  char* ws = (char*)d_ws;
  size_t off = 0;
  auto take = [&](size_t bytes) -> char* {
    off = (off + 255) & ~(size_t)255;
    char* p = ws + off;
    off += bytes;
    return p;
  };
  const size_t SPIKES = (size_t)TBX * NN * CC * sizeof(_Float16);  // 25.2 MB
  const size_t WTSZ   = (size_t)9 * CC * CC * sizeof(_Float16);    // 10.6 MB

  _Float16* xs = (_Float16*)take(SPIKES);
  _Float16* wtb[4];
  float *sc[4], *bi[4];
  for (int i = 0; i < 4; ++i) wtb[i] = (_Float16*)take(WTSZ);
  for (int i = 0; i < 4; ++i) {
    sc[i] = (float*)take(CC * sizeof(float));
    bi[i] = (float*)take(CC * sizeof(float));
  }
  float*    y    = (float*)take((size_t)TBX * CC * NN * sizeof(float)); // 50.3 MB
  _Float16* q_sp = (_Float16*)take(SPIKES);
  _Float16* k_sp = (_Float16*)take(SPIKES);
  _Float16* v_sp = (_Float16*)take(SPIKES);
  _Float16* o_sp = (_Float16*)take(SPIKES);

  for (int i = 0; i < 4; ++i) {
    const float* w  = (const float*)d_in[1 + 5 * i];
    const float* ga = (const float*)d_in[2 + 5 * i];
    const float* be = (const float*)d_in[3 + 5 * i];
    const float* mn = (const float*)d_in[4 + 5 * i];
    const float* vr = (const float*)d_in[5 + 5 * i];
    bnprep<<<dim3((CC + 255) / 256), dim3(256), 0, stream>>>(ga, be, mn, vr, sc[i], bi[i]);
    wtrans<<<dim3((CC * CC + 255) / 256), dim3(256), 0, stream>>>(w, wtb[i]);
  }

  const dim3 lifTgrid(CC / 32, NN / 32, BB);   // 24 x 8 x 16
  const dim3 convGrid(CC / 128, HH / 4, TBX);  // 6 x 4 x 64
  const int  lifPgrid = (int)(BCN / 256);      // 12288

  // head LIF: x -> xs spikes (TB,N,C)
  lif_transpose<<<lifTgrid, 256, 0, stream>>>(x, xs, 1.0f);

  // q path
  conv_bn_wmma<<<convGrid, 256, 0, stream>>>(xs, wtb[0], sc[0], bi[0], y);
  lif_transpose<<<lifTgrid, 256, 0, stream>>>(y, q_sp, 1.0f);
  // k path
  conv_bn_wmma<<<convGrid, 256, 0, stream>>>(xs, wtb[1], sc[1], bi[1], y);
  lif_plain<<<lifPgrid, 256, 0, stream>>>(y, k_sp, 1.0f);
  // v path
  conv_bn_wmma<<<convGrid, 256, 0, stream>>>(xs, wtb[2], sc[2], bi[2], y);
  lif_plain<<<lifPgrid, 256, 0, stream>>>(y, v_sp, 1.0f);

  // linear attention -> y (pre-LIF o, already scaled)
  attn_wmma<<<dim3(TBX * NHD), 256, 0, stream>>>(q_sp, k_sp, v_sp, y);

  // attn LIF (vth = 0.5) -> o spikes (TB,N,C)
  lif_transpose<<<lifTgrid, 256, 0, stream>>>(y, o_sp, 0.5f);

  // projection conv -> final output (T,B,C,H,W) f32
  conv_bn_wmma<<<convGrid, 256, 0, stream>>>(o_sp, wtb[3], sc[3], bi[3], (float*)d_out);
}
